// MSSS2D_82867099009411
// MI455X (gfx1250) — matrix operand order repaired and checked by reading, not compiled, not run on hardware
//
#include <hip/hip_runtime.h>
#include <hip/hip_bf16.h>

typedef __attribute__((ext_vector_type(16))) _Float16 v16h;
typedef __attribute__((ext_vector_type(8)))  float    v8f;

#define DE 384
#define DM 192
#define NB 4
#define LL 4096
#define L1 1024

__device__ __forceinline__ float silu_f(float v) { return v / (1.0f + __expf(-v)); }

// pack 8 f32 -> 8 f16 and store as one 16B LDS write (ds_store_b128)
__device__ __forceinline__ void cvt_store8(_Float16* dst, float4 a, float4 b) {
  union { _Float16 h[8]; float4 q; } u;
  u.h[0] = (_Float16)a.x; u.h[1] = (_Float16)a.y; u.h[2] = (_Float16)a.z; u.h[3] = (_Float16)a.w;
  u.h[4] = (_Float16)b.x; u.h[5] = (_Float16)b.y; u.h[6] = (_Float16)b.z; u.h[7] = (_Float16)b.w;
  *(float4*)dst = u.q;
}

// ---------------------------------------------------------------------------
// WMMA GEMM: O[m,n] = sum_k A[m,k] * W[n,k]   (A: MxK row-major, W: NxK row-major)
// Block tile 128(M) x 64(N), 8 waves, each wave 32x32 via 2x2 v_wmma_f32_16x16x32_f16.
// LDS row stride 40 halves (80B): 16B-aligned packed stores, conflict-free frag reads.
// EPI=0: in_proj epilogue (split into xh channel-first + silu(z));  EPI=1: plain store.
// ---------------------------------------------------------------------------
template <int EPI>
__global__ __launch_bounds__(256) void msss2d_gemm_wmma(
    const float* __restrict__ A, const float* __restrict__ W,
    float* __restrict__ O0, float* __restrict__ O1, int Kd)
{
  __shared__ _Float16 Alds[128 * 40];
  __shared__ _Float16 Blds[64 * 40];

  const int tid  = threadIdx.x;
  const int m0   = blockIdx.x * 128;
  const int n0   = blockIdx.y * 64;
  const int w    = tid >> 5, lane = tid & 31;
  const int wm   = (w & 3) * 32;
  const int wn   = (w >> 2) * 32;
  const int g    = lane >> 4;
  const int mr   = lane & 15;

  // 8-contiguous-float segments: A has 512 (2/thread), B has 256 (1/thread)
  const int rA0 = tid >> 2;
  const int rA1 = 64 + (tid >> 2);
  const int c8  = (tid & 3) << 3;

  v8f acc[2][2] = {};

  for (int k0 = 0; k0 < Kd; k0 += 32) {
    // phase 1: issue ALL global loads into distinct registers (MLP), then convert+store
    const float* pa0 = A + (size_t)(m0 + rA0) * Kd + k0 + c8;
    const float* pa1 = A + (size_t)(m0 + rA1) * Kd + k0 + c8;
    const float* pb  = W + (size_t)(n0 + rA0) * Kd + k0 + c8;   // rA0 in 0..63 here
    float4 a00 = *(const float4*)pa0;
    float4 a01 = *(const float4*)(pa0 + 4);
    float4 a10 = *(const float4*)pa1;
    float4 a11 = *(const float4*)(pa1 + 4);
    float4 b00 = *(const float4*)pb;
    float4 b01 = *(const float4*)(pb + 4);
    if (k0 + 32 < Kd)  // pull next K-slab toward cache (global_prefetch_b8)
      __builtin_prefetch(A + (size_t)(m0 + (tid >> 1)) * Kd + k0 + 32, 0, 1);
    cvt_store8(&Alds[rA0 * 40 + c8], a00, a01);
    cvt_store8(&Alds[rA1 * 40 + c8], a10, a11);
    cvt_store8(&Blds[rA0 * 40 + c8], b00, b01);
    __syncthreads();

    // B fragment (32x16 f16): lane holds N = wn+j*16+mr; e<8 -> K=8g+e, e>=8 -> K=16+8g+(e-8)
    v16h bf[2];
#pragma unroll
    for (int j = 0; j < 2; ++j) {
      const _Float16* bp = &Blds[(wn + j * 16 + mr) * 40];
#pragma unroll
      for (int e = 0; e < 8; ++e) { bf[j][e] = bp[8 * g + e]; bf[j][e + 8] = bp[16 + 8 * g + e]; }
    }
#pragma unroll
    for (int i = 0; i < 2; ++i) {
      // A fragment (16x32 f16): lane row M = wm+i*16+mr; e<8 -> K=8g+e, e>=8 -> K=16+8g+(e-8)
      const _Float16* ap = &Alds[(wm + i * 16 + mr) * 40];
      v16h af;
#pragma unroll
      for (int e = 0; e < 8; ++e) { af[e] = ap[8 * g + e]; af[e + 8] = ap[16 + 8 * g + e]; }
#pragma unroll
      for (int j = 0; j < 2; ++j)
        acc[i][j] = __builtin_amdgcn_wmma_f32_16x16x32_f16(
            false, af, false, bf[j], (short)0, acc[i][j], false, false);
    }
    __syncthreads();
  }

  // C/D layout: VGPR r, lane -> M = r + 8*g, N = mr
#pragma unroll
  for (int i = 0; i < 2; ++i)
#pragma unroll
    for (int j = 0; j < 2; ++j)
#pragma unroll
      for (int r = 0; r < 8; ++r) {
        int m = m0 + wm + i * 16 + r + 8 * g;
        int n = n0 + wn + j * 16 + mr;
        float v = acc[i][j][r];
        if (EPI == 0) {
          int b = m >> 12, hw = m & 4095;
          if (n < DE) O0[((size_t)b * DE + n) * LL + hw] = v;            // xh (B,DE,H,W)
          else        O1[(size_t)m * DE + (n - DE)]      = silu_f(v);    // z  (B*L,DE)
        } else {
          O0[(size_t)m * DM + n] = v;                                    // out (B*L,DM)
        }
      }
}

// ---------------------------------------------------------------------------
// Depthwise 3x3 conv, stride 1, pad 1, + SiLU.  layout (B,DE,64,64)
// ---------------------------------------------------------------------------
__global__ __launch_bounds__(256) void msss2d_conv3(
    const float* __restrict__ xh, const float* __restrict__ w,
    const float* __restrict__ bias, float* __restrict__ x0)
{
  int gid = blockIdx.x * 256 + threadIdx.x;
  int xx = gid & 63, yy = (gid >> 6) & 63;
  int t = gid >> 12;            // b*DE + c
  int c = t % DE;
  const float* src = xh + (size_t)t * LL;
  float acc = bias[c];
#pragma unroll
  for (int dy = 0; dy < 3; ++dy) {
    int iy = yy + dy - 1;
    if (iy < 0 || iy > 63) continue;
#pragma unroll
    for (int dx = 0; dx < 3; ++dx) {
      int ix = xx + dx - 1;
      if (ix < 0 || ix > 63) continue;
      acc += src[iy * 64 + ix] * w[c * 9 + dy * 3 + dx];
    }
  }
  x0[gid] = silu_f(acc);
}

// ---------------------------------------------------------------------------
// Depthwise 7x7 conv, stride 2, pad 3, + SiLU.  in (B,DE,64,64) -> out (B,DE,32,32)
// ---------------------------------------------------------------------------
__global__ __launch_bounds__(256) void msss2d_conv7(
    const float* __restrict__ xh, const float* __restrict__ w,
    const float* __restrict__ bias, float* __restrict__ xb1)
{
  int gid = blockIdx.x * 256 + threadIdx.x;
  int ox = gid & 31, oy = (gid >> 5) & 31;
  int t = gid >> 10;            // b*DE + c
  int c = t % DE;
  const float* src = xh + (size_t)t * LL;
  float acc = bias[c];
  for (int dy = 0; dy < 7; ++dy) {
    int iy = oy * 2 + dy - 3;
    if (iy < 0 || iy > 63) continue;
    for (int dx = 0; dx < 7; ++dx) {
      int ix = ox * 2 + dx - 3;
      if (ix < 0 || ix > 63) continue;
      acc += src[iy * 64 + ix] * w[c * 49 + dy * 7 + dx];
    }
  }
  xb1[gid] = silu_f(acc);
}

// scan-order index -> source index inside per-(b,d) plane
__device__ __forceinline__ int xs_src(int k, int l) {
  if (k == 0) return l;                       // full-res, forward
  if (k == 1) return (L1 - 1) - l;            // low-res, reversed
  int m = (k == 2) ? l : (L1 - 1) - l;        // transposed (and reversed for k=3)
  int w1 = m >> 5, h1 = m & 31;
  return h1 * 32 + w1;
}

// ---------------------------------------------------------------------------
// x_dbl projection: out[c,l] = sum_d xs[d,l] * pw[c,d], c in 0..43
// ---------------------------------------------------------------------------
__global__ __launch_bounds__(256) void msss2d_xproj(
    const float* __restrict__ xs, const float* __restrict__ pw,
    float* __restrict__ dts, float* __restrict__ bc, int k, int Lk)
{
  int gid = blockIdx.x * 256 + threadIdx.x;
  int l = gid % Lk;
  int c = (gid / Lk) % 44;
  int b = gid / (Lk * 44);
  int s = xs_src(k, l);
  const float* xp = xs + (size_t)b * DE * Lk + s;
  const float* wp = pw + c * DE;
  float acc = 0.f;
  for (int d = 0; d < DE; ++d) acc += xp[(size_t)d * Lk] * wp[d];
  if (c < 12) dts[((size_t)b * 12 + c) * Lk + l] = acc;
  else        bc[((size_t)b * 32 + (c - 12)) * Lk + l] = acc;
}

// ---------------------------------------------------------------------------
// dt[d,l] = softplus(sum_r dts[r,l]*dw[d,r] + db[d])
// ---------------------------------------------------------------------------
__global__ __launch_bounds__(256) void msss2d_dtproj(
    const float* __restrict__ dts, const float* __restrict__ dw,
    const float* __restrict__ db, float* __restrict__ dt, int Lk)
{
  int gid = blockIdx.x * 256 + threadIdx.x;
  int l = gid % Lk;
  int d = (gid / Lk) % DE;
  int b = gid / (Lk * DE);
  float acc = db[d];
  const float* dp = dts + (size_t)b * 12 * Lk + l;
#pragma unroll
  for (int r = 0; r < 12; ++r) acc += dp[(size_t)r * Lk] * dw[d * 12 + r];
  dt[((size_t)b * DE + d) * Lk + l] = (acc > 20.f) ? acc : log1pf(__expf(acc));
}

// ---------------------------------------------------------------------------
// Tensor Data Mover: 2D tile (tile_cols x tile_rows), 4B elements, row stride
// stride_elems, LDS destination padded +1 dword every 32 dwords (stride 33).
// D# per CDNA5 ISA 8.3/8.4: group0 = {flags, lds_addr, global_addr, type=2},
// group1 = {ctrl, dims/tiles/stride}, groups 2/3 (and trailing group) zero.
// This toolchain's builtin takes 6 args (g0 v4u, g1 v8i, g2 v4i, g3 v4i, v8i, cpol).
// ---------------------------------------------------------------------------
#if defined(__has_builtin)
#if __has_builtin(__builtin_amdgcn_tensor_load_to_lds) && __has_builtin(__builtin_amdgcn_s_wait_tensorcnt)
#define MSSS2D_HAS_TDM 1
#endif
#endif

#ifdef MSSS2D_HAS_TDM
typedef __attribute__((ext_vector_type(4))) unsigned int tdm_v4u;
typedef __attribute__((ext_vector_type(8))) int tdm_v8i;
typedef __attribute__((ext_vector_type(4))) int tdm_v4i;

__device__ __forceinline__ void tdm_load_2d(unsigned lds_byte, const float* gsrc,
                                            int tile_cols, int tile_rows, int stride_elems)
{
  unsigned long long ga = (unsigned long long)(size_t)(const void*)gsrc;
  tdm_v4u g0;
  g0[0] = 1u;                                                  // count=1, user mode
  g0[1] = lds_byte;                                            // lds_addr (bytes)
  g0[2] = (unsigned)ga;                                        // global_addr[31:0]
  g0[3] = (unsigned)((ga >> 32) & 0x01ffffffu) | (2u << 30);   // global_addr[56:32] | type=2
  tdm_v8i g1;
  g1[0] = (int)((2u << 16) | (1u << 20) | (4u << 22));         // data_size=4B, pad_en, interval=32dw, amount=1dw
  g1[1] = 0;                                                   // abar=0, tensor_dim0[15:0] (dim0=1<<30)
  g1[2] = 0x4000;                                              // tensor_dim0[31:16] ; tensor_dim1[15:0]=0
  g1[3] = (int)(0x4000u | ((unsigned)tile_cols << 16));        // tensor_dim1[31:16]=1<<30 | tile_dim0
  g1[4] = tile_rows & 0xffff;                                  // tile_dim1 ; tile_dim2=0
  g1[5] = stride_elems;                                        // tensor_dim0_stride[31:0]
  g1[6] = 0;
  g1[7] = 0;
  tdm_v4i gz4 = {0, 0, 0, 0};
  tdm_v8i gz8 = {0, 0, 0, 0, 0, 0, 0, 0};
  __builtin_amdgcn_tensor_load_to_lds(g0, g1, gz4, gz4, gz8, 0);
}
#endif

// ---------------------------------------------------------------------------
// Selective scan: one thread = one (b,d) recurrence, state h[16] in VGPRs.
// dt and B/C chunks are straight 2D strided copies -> TDM, double buffered
// (issue chunk i+1 before consuming chunk i, s_wait_tensorcnt(2) retires the
// oldest pair).  Permuted x reads stay as plain coalesced loads.
// ---------------------------------------------------------------------------
__global__ __launch_bounds__(128) void msss2d_scan(
    const float* __restrict__ xs, const float* __restrict__ dt,
    const float* __restrict__ bc, float* __restrict__ yout,
    const float* __restrict__ Alog, const float* __restrict__ Dsk,
    int k, int Lk)
{
  __shared__ float xls[128 * 33];
  __shared__ float dls[2][128 * 33 + 32];
  __shared__ float bcl[2][32 * 33 + 32];

  const int tid   = threadIdx.x;
  const int b     = blockIdx.x / 3;
  const int chunk = blockIdx.x % 3;
  const int d     = chunk * 128 + tid;
  const bool w0   = (tid >> 5) == 0;     // wave-uniform: TDM ignores EXEC, so gate by wave
  (void)w0;

  float Ae[16];
#pragma unroll
  for (int n = 0; n < 16; ++n) Ae[n] = -__expf(Alog[d * 16 + n]);
  const float Dv = Dsk[d];

  float h[16];
#pragma unroll
  for (int n = 0; n < 16; ++n) h[n] = 0.f;

  const float* xbase  = xs + (size_t)b * DE * Lk;
  const float* dbase  = dt + ((size_t)b * DE + chunk * 128) * Lk;
  const float* bcbase = bc + (size_t)b * 32 * Lk;
  float* ybase = yout + (size_t)b * Lk * DE + d;

  const int nc = Lk >> 5;
  int cur = 0;

#ifdef MSSS2D_HAS_TDM
  if (w0) {
    tdm_load_2d((unsigned)(size_t)(void*)&dls[0][0], dbase, 32, 128, Lk);
    tdm_load_2d((unsigned)(size_t)(void*)&bcl[0][0], bcbase, 32, 32, Lk);
  }
#endif

  for (int ci = 0; ci < nc; ++ci) {
    const int l0 = ci << 5;
    __syncthreads();                       // previous chunk fully consumed
    // stage x (direction-permuted gather)
#pragma unroll 4
    for (int j = 0; j < 32; ++j) {
      int idx = tid + j * 128;
      int lc = idx & 31, row = idx >> 5;
      xls[row * 33 + lc] = xbase[(size_t)(chunk * 128 + row) * Lk + xs_src(k, l0 + lc)];
    }
#ifdef MSSS2D_HAS_TDM
    if (w0) {
      if (ci + 1 < nc) {                   // prefetch next chunk via TDM, then retire current pair
        tdm_load_2d((unsigned)(size_t)(void*)&dls[cur ^ 1][0], dbase + (l0 + 32), 32, 128, Lk);
        tdm_load_2d((unsigned)(size_t)(void*)&bcl[cur ^ 1][0], bcbase + (l0 + 32), 32, 32, Lk);
        __builtin_amdgcn_s_wait_tensorcnt(2);
      } else {
        __builtin_amdgcn_s_wait_tensorcnt(0);
      }
    }
#else
    {
#pragma unroll 4
      for (int j = 0; j < 32; ++j) {
        int idx = tid + j * 128;
        int lc = idx & 31, row = idx >> 5;
        dls[cur][row * 33 + lc] = dbase[(size_t)row * Lk + (l0 + lc)];
      }
#pragma unroll
      for (int j = 0; j < 8; ++j) {
        int idx = tid + j * 128;
        int lc = idx & 31, n = idx >> 5;
        bcl[cur][n * 33 + lc] = bcbase[(size_t)n * Lk + (l0 + lc)];
      }
    }
#endif
    __syncthreads();

    const float* dc = &dls[cur][0];
    const float* bb = &bcl[cur][0];
    for (int s = 0; s < 32; ++s) {
      float xv  = xls[tid * 33 + s];
      float dtv = dc[tid * 33 + s];
      float dx = dtv * xv;
      float y = 0.f;
#pragma unroll
      for (int n = 0; n < 16; ++n) {
        h[n] = __expf(dtv * Ae[n]) * h[n] + dx * bb[n * 33 + s];
        y += h[n] * bb[(16 + n) * 33 + s];
      }
      ybase[(size_t)(l0 + s) * DE] = y + xv * Dv;
    }
    cur ^= 1;
  }
}

// ---------------------------------------------------------------------------
// LayerNorm over last dim (384) in place; one block per row.
// ---------------------------------------------------------------------------
__global__ __launch_bounds__(128) void msss2d_ln(
    float* __restrict__ y, const float* __restrict__ lw, const float* __restrict__ lb)
{
  __shared__ float s1[128], s2[128];
  float* row = y + (size_t)blockIdx.x * DE;
  const int tid = threadIdx.x;
  float v[3], sum = 0.f, sq = 0.f;
#pragma unroll
  for (int j = 0; j < 3; ++j) { v[j] = row[tid + j * 128]; sum += v[j]; sq += v[j] * v[j]; }
  s1[tid] = sum; s2[tid] = sq;
  __syncthreads();
  for (int off = 64; off > 0; off >>= 1) {
    if (tid < off) { s1[tid] += s1[tid + off]; s2[tid] += s2[tid + off]; }
    __syncthreads();
  }
  float mu  = s1[0] * (1.0f / DE);
  float var = s2[0] * (1.0f / DE) - mu * mu;
  float inv = rsqrtf(var + 1e-5f);
#pragma unroll
  for (int j = 0; j < 3; ++j) {
    int e = tid + j * 128;
    row[e] = (v[j] - mu) * inv * lw[e] + lb[e];
  }
}

// three scan directions mapped back to the (h1,w1) low-res grid
__device__ __forceinline__ float yb1_at(const float* __restrict__ Y1,
                                        int b, int h1, int w1, int d)
{
  int lf = h1 * 32 + w1;
  int lt = w1 * 32 + h1;
  size_t bb = (size_t)b * L1 * DE;
  size_t pk = (size_t)NB * L1 * DE;
  float a = Y1[bb + (size_t)(1023 - lf) * DE + d];            // hori reversed
  float c = Y1[pk + bb + (size_t)lt * DE + d];                // vertical
  float e = Y1[2 * pk + bb + (size_t)(1023 - lt) * DE + d];   // vertical reversed
  return a + c + e;
}

// ---------------------------------------------------------------------------
// Bilinear upsample (half-pixel centers) of low-res branch, add y0, gate with z.
// ---------------------------------------------------------------------------
__global__ __launch_bounds__(256) void msss2d_combine(
    const float* __restrict__ Y0p, const float* __restrict__ Y1p,
    const float* __restrict__ Zp, float* __restrict__ YCp)
{
  int gid = blockIdx.x * 256 + threadIdx.x;
  int d = gid % DE;
  int bl = gid / DE;                 // b*4096 + l
  int l = bl & 4095;
  int b = bl >> 12;
  int yy = l >> 6, xx = l & 63;
  float fy = yy * 0.5f - 0.25f, fx = xx * 0.5f - 0.25f;
  int iy0 = (int)floorf(fy), ix0 = (int)floorf(fx);
  float wy = fy - (float)iy0, wx = fx - (float)ix0;
  int iy1 = (iy0 + 1 > 31) ? 31 : iy0 + 1;
  int ix1 = (ix0 + 1 > 31) ? 31 : ix0 + 1;
  iy0 = (iy0 < 0) ? 0 : iy0;
  ix0 = (ix0 < 0) ? 0 : ix0;
  float v00 = yb1_at(Y1p, b, iy0, ix0, d);
  float v01 = yb1_at(Y1p, b, iy0, ix1, d);
  float v10 = yb1_at(Y1p, b, iy1, ix0, d);
  float v11 = yb1_at(Y1p, b, iy1, ix1, d);
  float up = (v00 * (1.f - wx) + v01 * wx) * (1.f - wy)
           + (v10 * (1.f - wx) + v11 * wx) * wy;
  size_t o = (size_t)bl * DE + d;
  YCp[o] = (Y0p[o] + up) * Zp[o];
}

// ---------------------------------------------------------------------------
extern "C" void kernel_launch(void* const* d_in, const int* in_sizes, int n_in,
                              void* d_out, int out_size, void* d_ws, size_t ws_size,
                              hipStream_t stream)
{
  (void)in_sizes; (void)n_in; (void)out_size; (void)ws_size;
  const float* x       = (const float*)d_in[0];
  const float* in_w    = (const float*)d_in[1];
  const float* conv_w  = (const float*)d_in[2];
  const float* conv_b  = (const float*)d_in[3];
  const float* conv1_w = (const float*)d_in[4];
  const float* conv1_b = (const float*)d_in[5];
  const float* xpw     = (const float*)d_in[6];
  const float* dtw     = (const float*)d_in[7];
  const float* dtbias  = (const float*)d_in[8];
  const float* alog    = (const float*)d_in[9];
  const float* dsk     = (const float*)d_in[10];
  const float* lnw     = (const float*)d_in[11];
  const float* lnb     = (const float*)d_in[12];
  const float* outw    = (const float*)d_in[13];
  float* out = (float*)d_out;
  float* ws  = (float*)d_ws;

  // workspace layout (floats)
  const size_t XH   = 0;
  const size_t Z    = XH   + (size_t)NB * DE * LL;
  const size_t X0   = Z    + (size_t)NB * LL * DE;
  const size_t XB1  = X0   + (size_t)NB * DE * LL;
  const size_t DTS0 = XB1  + (size_t)NB * DE * L1;
  const size_t DTS1 = DTS0 + (size_t)NB * 12 * LL;
  const size_t BC0  = DTS1 + (size_t)3 * NB * 12 * L1;
  const size_t BC1  = BC0  + (size_t)NB * 32 * LL;
  const size_t DT0  = BC1  + (size_t)3 * NB * 32 * L1;
  const size_t DT1  = DT0  + (size_t)NB * DE * LL;
  const size_t Y0   = DT1  + (size_t)3 * NB * DE * L1;
  const size_t Y1   = Y0   + (size_t)NB * LL * DE;   // contiguous after Y0 (single LN pass)
  const size_t YC   = XH;                            // alias: xh dead after convs

  // 1. in_proj (WMMA): (16384x192) x (192x768) -> xh + silu(z)
  msss2d_gemm_wmma<0><<<dim3(128, 12), 256, 0, stream>>>(x, in_w, ws + XH, ws + Z, DM);

  // 2. depthwise conv branches
  msss2d_conv3<<<(NB * DE * LL) / 256, 256, 0, stream>>>(ws + XH, conv_w, conv_b, ws + X0);
  msss2d_conv7<<<(NB * DE * L1) / 256, 256, 0, stream>>>(ws + XH, conv1_w, conv1_b, ws + XB1);

  // 3-5. per scan direction: x_dbl projection, dt projection, selective scan
  for (int k = 0; k < 4; ++k) {
    const int Lk = k ? L1 : LL;
    const float* xsrc = k ? (ws + XB1) : (ws + X0);
    float* dts = k ? (ws + DTS1 + (size_t)(k - 1) * NB * 12 * L1) : (ws + DTS0);
    float* bcp = k ? (ws + BC1  + (size_t)(k - 1) * NB * 32 * L1) : (ws + BC0);
    float* dtp = k ? (ws + DT1  + (size_t)(k - 1) * NB * DE * L1) : (ws + DT0);
    float* yb  = k ? (ws + Y1   + (size_t)(k - 1) * NB * L1 * DE) : (ws + Y0);
    msss2d_xproj<<<(NB * 44 * Lk) / 256, 256, 0, stream>>>(
        xsrc, xpw + (size_t)k * 44 * DE, dts, bcp, k, Lk);
    msss2d_dtproj<<<(NB * DE * Lk) / 256, 256, 0, stream>>>(
        dts, dtw + (size_t)k * DE * 12, dtbias + (size_t)k * DE, dtp, Lk);
    msss2d_scan<<<NB * 3, 128, 0, stream>>>(
        xsrc, dtp, bcp, yb, alog + (size_t)k * DE * 16, dsk + (size_t)k * DE, k, Lk);
  }

  // 6. LayerNorm: Y0 rows (B*L) and Y1 rows (3*B*L1) are contiguous
  msss2d_ln<<<NB * LL + 3 * NB * L1, 128, 0, stream>>>(ws + Y0, lnw, lnb);

  // 7. upsample + add + gate
  msss2d_combine<<<(NB * LL * DE) / 256, 256, 0, stream>>>(
      ws + Y0, ws + Y1, ws + Z, ws + YC);

  // 8. out_proj (WMMA): (16384x384) x (384x192) -> output
  msss2d_gemm_wmma<1><<<dim3(128, 3), 256, 0, stream>>>(ws + YC, outw, out, nullptr, DE);
}
